// IPMPDenoiser_7627861918049
// MI455X (gfx1250) — compile-verified, matrix operand order checked by reading, not run on hardware
//
#include <hip/hip_runtime.h>
#include <hip/hip_bf16.h>

typedef _Float16 v16h __attribute__((ext_vector_type(16)));
typedef _Float16 v8h  __attribute__((ext_vector_type(8)));
typedef float    v8f  __attribute__((ext_vector_type(8)));

#define N_NODE 2048
#define K_NN   30
#define N_EDGE (N_NODE * K_NN)

// ---------------- WMMA fragment helpers (gfx1250 wave32) ----------------
// A tile (16x32 f16) layout per ISA: lane l holds row m=(l&15); halves h<8 ->
// k = k0 + h + ((l>>4)<<3); h>=8 -> k = k0 + 16 + (h-8) + ((l>>4)<<3).
__device__ __forceinline__ v16h load_a_frag(const _Float16* aT, int ldk, int mt, int k0) {
  const int lane = threadIdx.x & 31;
  const int m  = mt * 16 + (lane & 15);
  const int kb = k0 + ((lane >> 4) << 3);
  const _Float16* p = aT + m * ldk + kb;
  v8h lo = *(const v8h*)(p);        // k = kb .. kb+7
  v8h hi = *(const v8h*)(p + 16);   // k = kb+16 .. kb+23
  v16h r;
#pragma unroll
  for (int i = 0; i < 8; ++i) { r[i] = lo[i]; r[8 + i] = hi[i]; }
  return r;
}

// B fragments are pre-packed by conv_wfrag_kernel so each lane loads 16
// contiguous halves (32 bytes) per 32x16 K-tile.
__device__ __forceinline__ v16h load_b_frag(const _Float16* wf, int ntiles, int kt, int nt) {
  const int lane = threadIdx.x & 31;
  return *(const v16h*)(wf + (((size_t)(kt * ntiles + nt)) << 9) + (lane << 4));
}

// 32-row tile GEMM: C[32 x dout] = A[32 x 32*kChunks] * Wfrag + bias, optional relu.
__device__ __forceinline__ void gemm32_f16out(const _Float16* aT, int ldk, int kChunks,
                                              const _Float16* wf, const float* bias,
                                              int dout, _Float16* outT, int relu) {
  const int wave = threadIdx.x >> 5;
  const int lane = threadIdx.x & 31;
  const int nWaves = blockDim.x >> 5;
  const int ntiles = dout >> 4;
  for (int tile = wave; tile < 2 * ntiles; tile += nWaves) {
    const int mt = tile & 1, nt = tile >> 1;
    v8f c = {};
    for (int kc = 0; kc < kChunks; ++kc) {
      v16h a = load_a_frag(aT, ldk, mt, kc * 32);
      v16h b = load_b_frag(wf, ntiles, kc, nt);
      c = __builtin_amdgcn_wmma_f32_16x16x32_f16(false, a, false, b, (short)0, c, false, false);
    }
    const int n  = nt * 16 + (lane & 15);
    const int mb = mt * 16 + ((lane >> 4) << 3);
    const float bv = bias ? bias[n] : 0.f;
#pragma unroll
    for (int r = 0; r < 8; ++r) {
      float v = c[r] + bv;
      if (relu) v = v > 0.f ? v : 0.f;
      outT[(mb + r) * dout + n] = (_Float16)v;
    }
  }
}

__device__ __forceinline__ void gemm32_f32out(const _Float16* aT, int ldk, int kChunks,
                                              const _Float16* wf, const float* bias,
                                              int dout, float* outT) {
  const int wave = threadIdx.x >> 5;
  const int lane = threadIdx.x & 31;
  const int nWaves = blockDim.x >> 5;
  const int ntiles = dout >> 4;
  for (int tile = wave; tile < 2 * ntiles; tile += nWaves) {
    const int mt = tile & 1, nt = tile >> 1;
    v8f c = {};
    for (int kc = 0; kc < kChunks; ++kc) {
      v16h a = load_a_frag(aT, ldk, mt, kc * 32);
      v16h b = load_b_frag(wf, ntiles, kc, nt);
      c = __builtin_amdgcn_wmma_f32_16x16x32_f16(false, a, false, b, (short)0, c, false, false);
    }
    const int n  = nt * 16 + (lane & 15);
    const int mb = mt * 16 + ((lane >> 4) << 3);
    const float bv = bias ? bias[n] : 0.f;
#pragma unroll
    for (int r = 0; r < 8; ++r) outT[(mb + r) * dout + n] = c[r] + bv;
  }
}

// LayerNorm over rows of a [32 x d] f32 LDS tile (one wave per row, strided).
__device__ __forceinline__ void ln_rows(float* T, int d, const float* g, const float* b) {
  const int wave = threadIdx.x >> 5, lane = threadIdx.x & 31;
  const int nWaves = blockDim.x >> 5;
  for (int m = wave; m < 32; m += nWaves) {
    float s = 0.f;
    for (int k = lane; k < d; k += 32) s += T[m * d + k];
    for (int off = 16; off; off >>= 1) s += __shfl_xor(s, off, 32);
    const float mu = s / (float)d;
    float v = 0.f;
    for (int k = lane; k < d; k += 32) { float x = T[m * d + k] - mu; v += x * x; }
    for (int off = 16; off; off >>= 1) v += __shfl_xor(v, off, 32);
    const float rs = rsqrtf(v / (float)d + 1e-5f);
    for (int k = lane; k < d; k += 32) T[m * d + k] = (T[m * d + k] - mu) * rs * g[k] + b[k];
  }
}

// ---------------- weight repack: f32 row-major -> f16 B-fragment order ----------------
__global__ void conv_wfrag_kernel(const float* __restrict__ srcW, _Float16* __restrict__ dst,
                                  int din, int dinPad, int dout) {
  const size_t total = (size_t)dinPad * dout;
  const int ntiles = dout >> 4;
  for (size_t idx = blockIdx.x * (size_t)blockDim.x + threadIdx.x; idx < total;
       idx += (size_t)gridDim.x * blockDim.x) {
    const int h = (int)(idx & 15);
    const int lane = (int)((idx >> 4) & 31);
    const size_t blk = idx >> 9;
    const int nt = (int)(blk % ntiles);
    const int kt = (int)(blk / ntiles);
    const int n = nt * 16 + (lane & 15);
    const int k = kt * 32 + ((lane >> 4) << 3) + (h < 8 ? h : (h - 8) + 16);
    float v = (k < din) ? srcW[(size_t)k * dout + n] : 0.f;
    dst[idx] = (_Float16)v;
  }
}

// ---------------- kNN (top-30 smallest d2, index tie-break) ----------------
__global__ void knn_kernel(const float* __restrict__ xca, int* __restrict__ srcIdx) {
  const int i = blockIdx.x * blockDim.x + threadIdx.x;
  if (i >= N_NODE) return;
  float bd[K_NN]; int bi[K_NN];
#pragma unroll
  for (int k = 0; k < K_NN; ++k) { bd[k] = 3.0e38f; bi[k] = 0; }
  const float xi = xca[i * 3 + 0], yi = xca[i * 3 + 1], zi = xca[i * 3 + 2];
  for (int j = 0; j < N_NODE; ++j) {
    if (j == i) continue;
    const float dx = xca[j * 3 + 0] - xi, dy = xca[j * 3 + 1] - yi, dz = xca[j * 3 + 2] - zi;
    const float d2 = dx * dx + dy * dy + dz * dz;
    if (d2 < bd[K_NN - 1]) {
      int p = K_NN - 1;
      while (p > 0 && bd[p - 1] > d2) { bd[p] = bd[p - 1]; bi[p] = bi[p - 1]; --p; }
      bd[p] = d2; bi[p] = j;
    }
  }
  for (int k = 0; k < K_NN; ++k) srcIdx[i * K_NN + k] = bi[k];
}

// ---------------- node embedding: 203->512->512->256 + LN ----------------
__global__ void node_embed_kernel(const float* __restrict__ nsin, const float* __restrict__ mask,
                                  const _Float16* wf0, const float* b0,
                                  const _Float16* wf1, const float* b1,
                                  const _Float16* wf2, const float* b2,
                                  const float* lng, const float* lnb,
                                  float* __restrict__ nodeOut) {
  extern __shared__ char smem[];
  _Float16* A0 = (_Float16*)smem;            // 32 x 224
  _Float16* H1 = A0 + 32 * 224;              // 32 x 512
  _Float16* H2 = H1 + 32 * 512;              // 32 x 512
  float*    O  = (float*)(H2 + 32 * 512);    // 32 x 256
  const int n0 = blockIdx.x * 32;
  for (int idx = threadIdx.x; idx < 32 * 224; idx += blockDim.x) {
    const int r = idx / 224, k = idx % 224;
    float v = (k < 203) ? nsin[(size_t)(n0 + r) * 203 + k] * mask[n0 + r] : 0.f;
    A0[idx] = (_Float16)v;
  }
  __syncthreads();
  gemm32_f16out(A0, 224, 7, wf0, b0, 512, H1, 1);
  __syncthreads();
  gemm32_f16out(H1, 512, 16, wf1, b1, 512, H2, 1);
  __syncthreads();
  gemm32_f32out(H2, 512, 16, wf2, b2, 256, O);
  __syncthreads();
  ln_rows(O, 256, lng, lnb);
  __syncthreads();
  for (int idx = threadIdx.x; idx < 32 * 256; idx += blockDim.x)
    nodeOut[(size_t)n0 * 256 + idx] = O[idx];
}

// ---------------- edge embedding: RBF(26 dists)=416 -> 256 -> 256 -> 128 + LN ----------------
__global__ void edge_embed_kernel(const int* __restrict__ srcIdx,
                                  const float* __restrict__ xca, const float* __restrict__ atoms,
                                  const float* __restrict__ mask,
                                  const _Float16* wf0, const float* b0,
                                  const _Float16* wf1, const float* b1,
                                  const _Float16* wf2, const float* b2,
                                  const float* lng, const float* lnb,
                                  float* __restrict__ zB) {
  extern __shared__ char smem[];
  _Float16* A  = (_Float16*)smem;            // 32 x 416
  _Float16* H1 = A + 32 * 416;               // 32 x 256
  _Float16* H2 = H1 + 32 * 256;              // 32 x 256
  float*    O  = (float*)(H2 + 32 * 256);    // 32 x 128
  float*    D  = O + 32 * 128;               // 32 x 26 distances
  const int e0 = blockIdx.x * 32;
  for (int idx = threadIdx.x; idx < 32 * 26; idx += blockDim.x) {
    const int r = idx / 26, a = idx % 26;
    const int e = e0 + r;
    const int s = srcIdx[e], dn = e / K_NN;
    float d2 = 0.f;
    if (a < 25) {
      const int si = a / 5, dj = a % 5;
#pragma unroll
      for (int c = 0; c < 3; ++c) {
        float dd = atoms[(size_t)(s * 5 + si) * 3 + c] - atoms[(size_t)(dn * 5 + dj) * 3 + c];
        d2 += dd * dd;
      }
    } else {
#pragma unroll
      for (int c = 0; c < 3; ++c) {
        float dd = xca[s * 3 + c] - xca[dn * 3 + c];
        d2 += dd * dd;
      }
    }
    D[idx] = sqrtf(d2 + 1e-8f);
  }
  __syncthreads();
  for (int idx = threadIdx.x; idx < 32 * 416; idx += blockDim.x) {
    const int r = idx / 416, k = idx % 416;
    const int a = k >> 4, q = k & 15;
    const int e = e0 + r;
    const float cq = 20.f * (float)q / 15.f;
    const float u = (D[r * 26 + a] - cq) * 0.8f;  // /1.25
    const float mprod = mask[srcIdx[e]] * mask[e / K_NN];
    A[idx] = (_Float16)(expf(-u * u) * mprod);
  }
  __syncthreads();
  gemm32_f16out(A, 416, 13, wf0, b0, 256, H1, 1);
  __syncthreads();
  gemm32_f16out(H1, 256, 8, wf1, b1, 256, H2, 1);
  __syncthreads();
  gemm32_f32out(H2, 256, 8, wf2, b2, 128, O);
  __syncthreads();
  ln_rows(O, 128, lng, lnb);
  __syncthreads();
  for (int idx = threadIdx.x; idx < 32 * 128; idx += blockDim.x)
    zB[(size_t)e0 * 128 + idx] = O[idx];
}

// ---------------- per-layer: points q = s@ptw+b ; pts = R q + t ----------------
__global__ void points_kernel(const float* __restrict__ node, const float* __restrict__ lat,
                              const float* __restrict__ ptw, const float* __restrict__ ptb,
                              const float* __restrict__ Rm, const float* __restrict__ tv,
                              float* __restrict__ pts) {
  __shared__ float q[8][12];
  const int local = threadIdx.x >> 4;
  const int j = threadIdx.x & 15;
  const int nn = blockIdx.x * 8 + local;
  if (j < 12) {
    float acc = ptb[j];
    for (int k = 0; k < 256; ++k) acc += node[(size_t)nn * 256 + k] * ptw[k * 12 + j];
    for (int k = 0; k < 128; ++k) acc += lat[(size_t)nn * 128 + k] * ptw[(256 + k) * 12 + j];
    q[local][j] = acc;
  }
  __syncthreads();
  if (j < 12) {
    const int p = j / 3, ii = j % 3;
    float v = tv[nn * 3 + ii];
#pragma unroll
    for (int jj = 0; jj < 3; ++jj) v += Rm[nn * 9 + ii * 3 + jj] * q[local][p * 3 + jj];
    pts[nn * 12 + p * 3 + ii] = v;
  }
}

// ---------------- per-layer: messages msg1->relu->msg2, aggregated per dst node ----------------
__global__ void msg_kernel(const int* __restrict__ srcIdx, const float* __restrict__ node,
                           const float* __restrict__ lat, const float* __restrict__ zB,
                           const float* __restrict__ ptsB, const float* __restrict__ Rm,
                           const _Float16* wfm1, const float* m1b,
                           const _Float16* wfm2, const float* m2b,
                           float* __restrict__ agg) {
  extern __shared__ char smem[];
  _Float16* A = (_Float16*)smem;            // 32 x 1184 (rows 30,31 zero)
  _Float16* H = A + 32 * 1184;              // 32 x 256
  float* dl   = (float*)(H + 32 * 256);     // 32 x 16 : dloc[12] + dist[4]
  float* aggL = dl + 32 * 16;               // 512
  const int i = blockIdx.x;
  for (int idx = threadIdx.x; idx < 30 * 4; idx += blockDim.x) {
    const int r = idx >> 2, p = idx & 3;
    const int e = i * K_NN + r;
    const int s = srcIdx[e];
    const float d0 = ptsB[s * 12 + p * 3 + 0] - ptsB[i * 12 + p * 3 + 0];
    const float d1 = ptsB[s * 12 + p * 3 + 1] - ptsB[i * 12 + p * 3 + 1];
    const float d2 = ptsB[s * 12 + p * 3 + 2] - ptsB[i * 12 + p * 3 + 2];
    // dloc_i = sum_j R[dst][j][i] * d_j  (R^T d)
    const float l0 = Rm[i * 9 + 0] * d0 + Rm[i * 9 + 3] * d1 + Rm[i * 9 + 6] * d2;
    const float l1 = Rm[i * 9 + 1] * d0 + Rm[i * 9 + 4] * d1 + Rm[i * 9 + 7] * d2;
    const float l2 = Rm[i * 9 + 2] * d0 + Rm[i * 9 + 5] * d1 + Rm[i * 9 + 8] * d2;
    dl[r * 16 + p * 3 + 0] = l0;
    dl[r * 16 + p * 3 + 1] = l1;
    dl[r * 16 + p * 3 + 2] = l2;
    dl[r * 16 + 12 + p] = sqrtf(l0 * l0 + l1 * l1 + l2 * l2 + 1e-8f);
  }
  __syncthreads();
  for (int idx = threadIdx.x; idx < 32 * 1184; idx += blockDim.x) {
    const int r = idx / 1184, k = idx % 1184;
    float v = 0.f;
    if (r < 30 && k < 1168) {
      const int e = i * K_NN + r;
      if (k < 256)        v = node[(size_t)srcIdx[e] * 256 + k];
      else if (k < 384)   v = lat[(size_t)srcIdx[e] * 128 + (k - 256)];
      else if (k < 512)   v = 0.f;
      else if (k < 768)   v = node[(size_t)i * 256 + (k - 512)];
      else if (k < 896)   v = lat[(size_t)i * 128 + (k - 768)];
      else if (k < 1024)  v = 0.f;
      else if (k < 1152)  v = zB[(size_t)e * 128 + (k - 1024)];
      else                v = dl[r * 16 + (k - 1152)];
    }
    A[idx] = (_Float16)v;
  }
  __syncthreads();
  gemm32_f16out(A, 1184, 37, wfm1, m1b, 256, H, 1);
  __syncthreads();
  for (int kk = threadIdx.x; kk < 512; kk += blockDim.x) aggL[kk] = 0.f;
  __syncthreads();
  {  // msg2 GEMM fused with masked column sum over the 30 valid edge rows
    const int wave = threadIdx.x >> 5, lane = threadIdx.x & 31;
    const int nWaves = blockDim.x >> 5;
    for (int tile = wave; tile < 64; tile += nWaves) {
      const int mt = tile & 1, nt = tile >> 1;
      v8f c = {};
      for (int kc = 0; kc < 8; ++kc) {
        v16h a = load_a_frag(H, 256, mt, kc * 32);
        v16h b = load_b_frag(wfm2, 32, kc, nt);
        c = __builtin_amdgcn_wmma_f32_16x16x32_f16(false, a, false, b, (short)0, c, false, false);
      }
      const int n  = nt * 16 + (lane & 15);
      const int mb = mt * 16 + ((lane >> 4) << 3);
      const float bv = m2b[n];
      float part = 0.f;
#pragma unroll
      for (int r = 0; r < 8; ++r)
        if (mb + r < 30) part += c[r] + bv;
      atomicAdd(&aggL[n], part);
    }
  }
  __syncthreads();
  for (int kk = threadIdx.x; kk < 512; kk += blockDim.x)
    agg[(size_t)i * 512 + kk] = aggL[kk] * (1.0f / (float)K_NN);
}

// ---------------- per-layer: edge update e1->relu->e2 + residual + LN ----------------
__global__ void edge_upd_kernel(const int* __restrict__ srcIdx, const float* __restrict__ node,
                                const float* __restrict__ lat, float* __restrict__ zB,
                                const _Float16* wfe1, const float* e1b,
                                const _Float16* wfe2, const float* e2b,
                                const float* lng, const float* lnb) {
  extern __shared__ char smem[];
  _Float16* A = (_Float16*)smem;            // 32 x 1152
  _Float16* H = A + 32 * 1152;              // 32 x 256
  float*    O = (float*)(H + 32 * 256);     // 32 x 128
  const int e0 = blockIdx.x * 32;
  for (int idx = threadIdx.x; idx < 32 * 1152; idx += blockDim.x) {
    const int r = idx / 1152, k = idx % 1152;
    const int e = e0 + r;
    const int dn = e / K_NN;
    float v;
    if (k < 256)       v = node[(size_t)srcIdx[e] * 256 + k];
    else if (k < 384)  v = lat[(size_t)srcIdx[e] * 128 + (k - 256)];
    else if (k < 512)  v = 0.f;
    else if (k < 768)  v = node[(size_t)dn * 256 + (k - 512)];
    else if (k < 896)  v = lat[(size_t)dn * 128 + (k - 768)];
    else if (k < 1024) v = 0.f;
    else               v = zB[(size_t)e * 128 + (k - 1024)];
    A[idx] = (_Float16)v;
  }
  __syncthreads();
  gemm32_f16out(A, 1152, 36, wfe1, e1b, 256, H, 1);
  __syncthreads();
  gemm32_f32out(H, 256, 8, wfe2, e2b, 128, O);
  __syncthreads();
  for (int idx = threadIdx.x; idx < 32 * 128; idx += blockDim.x)
    O[idx] += zB[(size_t)e0 * 128 + idx];
  __syncthreads();
  ln_rows(O, 128, lng, lnb);
  __syncthreads();
  for (int idx = threadIdx.x; idx < 32 * 128; idx += blockDim.x)
    zB[(size_t)e0 * 128 + idx] = O[idx];
}

// ---------------- per-layer: joint = LN(s + agg@Wj)*mask ; latent += joint@Wl ; node = LN(node+joint@Wn) ----
__global__ void joint_kernel(const float* __restrict__ agg, float* __restrict__ node,
                             float* __restrict__ lat, const float* __restrict__ mask,
                             const _Float16* wfJ, const float* Jb, const float* jlng, const float* jlnb,
                             const _Float16* wfL, const float* Lb,
                             const _Float16* wfN, const float* Nb, const float* nlng, const float* nlnb) {
  extern __shared__ char smem[];
  float*    sN = (float*)smem;                 // 32 x 256
  float*    sL = sN + 32 * 256;                // 32 x 128
  _Float16* Ah = (_Float16*)(sL + 32 * 128);   // 32 x 512 (agg f16, later joint f16)
  float*    J  = (float*)(Ah + 32 * 512);      // 32 x 512
  const int n0 = blockIdx.x * 32;
  for (int idx = threadIdx.x; idx < 32 * 256; idx += blockDim.x) sN[idx] = node[(size_t)n0 * 256 + idx];
  for (int idx = threadIdx.x; idx < 32 * 128; idx += blockDim.x) sL[idx] = lat[(size_t)n0 * 128 + idx];
  for (int idx = threadIdx.x; idx < 32 * 512; idx += blockDim.x)
    Ah[idx] = (_Float16)agg[(size_t)n0 * 512 + idx];
  __syncthreads();
  gemm32_f32out(Ah, 512, 16, wfJ, Jb, 512, J);
  __syncthreads();
  for (int idx = threadIdx.x; idx < 32 * 512; idx += blockDim.x) {
    const int r = idx >> 9, c = idx & 511;
    const float s = (c < 256) ? sN[r * 256 + c] : (c < 384 ? sL[r * 128 + (c - 256)] : 0.f);
    J[idx] += s;
  }
  __syncthreads();
  ln_rows(J, 512, jlng, jlnb);
  __syncthreads();
  for (int idx = threadIdx.x; idx < 32 * 512; idx += blockDim.x) {
    const int r = idx >> 9;
    Ah[idx] = (_Float16)(J[idx] * mask[n0 + r]);   // joint, masked
  }
  __syncthreads();
  gemm32_f32out(Ah, 512, 16, wfL, Lb, 128, J);     // latent delta
  __syncthreads();
  for (int idx = threadIdx.x; idx < 32 * 128; idx += blockDim.x)
    lat[(size_t)n0 * 128 + idx] = sL[idx] + J[idx];
  __syncthreads();
  gemm32_f32out(Ah, 512, 16, wfN, Nb, 256, J);     // node delta
  __syncthreads();
  for (int idx = threadIdx.x; idx < 32 * 256; idx += blockDim.x) J[idx] += sN[idx];
  __syncthreads();
  ln_rows(J, 256, nlng, nlnb);
  __syncthreads();
  for (int idx = threadIdx.x; idx < 32 * 256; idx += blockDim.x)
    node[(size_t)n0 * 256 + idx] = J[idx];
}

// ============================== host ==============================
struct LayerP {
  const float *pt_w, *pt_b, *msg1_w, *msg1_b, *msg2_w, *msg2_b;
  const float *joint_w, *joint_b, *jln_g, *jln_b;
  const float *e1_w, *e1_b, *e2_w, *e2_b, *eln_g, *eln_b;
  const float *lat_w, *lat_b, *nd_w, *nd_b, *nln_g, *nln_b;
};

extern "C" void kernel_launch(void* const* d_in, const int* in_sizes, int n_in,
                              void* d_out, int out_size, void* d_ws, size_t ws_size,
                              hipStream_t stream) {
  if (n_in < 111) return;
  auto g = [&](int i) { return (const float*)d_in[i]; };
  const float* xca   = g(0);
  const float* atoms = g(1);
  const float* nsin  = g(2);
  const float* lat0  = g(3);
  const float* Rm    = g(4);
  const float* tv    = g(5);
  const float* mask  = g(6);

  const float *en_w[3], *en_b[3], *enln_g, *enln_b;
  const float *ee_w[3], *ee_b[3], *eeln_g, *eeln_b;
  LayerP L[4];

  if (in_sizes[7] == 128) {
    // JAX tree order (dict keys sorted)
    eeln_b = g(7);  eeln_g = g(8);
    ee_b[0] = g(9);  ee_w[0] = g(10);
    ee_b[1] = g(11); ee_w[1] = g(12);
    ee_b[2] = g(13); ee_w[2] = g(14);
    en_b[0] = g(15); en_w[0] = g(16);
    en_b[1] = g(17); en_w[1] = g(18);
    en_b[2] = g(19); en_w[2] = g(20);
    enln_b = g(21); enln_g = g(22);
    for (int i = 0; i < 4; ++i) {
      const int B = 23 + i * 22;
      L[i].e1_b = g(B + 0);  L[i].e1_w = g(B + 1);
      L[i].e2_b = g(B + 2);  L[i].e2_w = g(B + 3);
      L[i].eln_b = g(B + 4); L[i].eln_g = g(B + 5);
      L[i].jln_b = g(B + 6); L[i].jln_g = g(B + 7);
      L[i].joint_b = g(B + 8); L[i].joint_w = g(B + 9);
      L[i].lat_b = g(B + 10); L[i].lat_w = g(B + 11);
      L[i].msg1_b = g(B + 12); L[i].msg1_w = g(B + 13);
      L[i].msg2_b = g(B + 14); L[i].msg2_w = g(B + 15);
      L[i].nd_b = g(B + 16); L[i].nd_w = g(B + 17);
      L[i].nln_b = g(B + 18); L[i].nln_g = g(B + 19);
      L[i].pt_b = g(B + 20); L[i].pt_w = g(B + 21);
    }
  } else if (in_sizes[7] == 203 * 512) {
    // dict insertion order, w before b, g before b
    en_w[0] = g(7);  en_b[0] = g(8);
    en_w[1] = g(9);  en_b[1] = g(10);
    en_w[2] = g(11); en_b[2] = g(12);
    enln_g = g(13); enln_b = g(14);
    ee_w[0] = g(15); ee_b[0] = g(16);
    ee_w[1] = g(17); ee_b[1] = g(18);
    ee_w[2] = g(19); ee_b[2] = g(20);
    eeln_g = g(21); eeln_b = g(22);
    for (int i = 0; i < 4; ++i) {
      const int B = 23 + i * 22;
      L[i].pt_w = g(B + 0);  L[i].pt_b = g(B + 1);
      L[i].msg1_w = g(B + 2); L[i].msg1_b = g(B + 3);
      L[i].msg2_w = g(B + 4); L[i].msg2_b = g(B + 5);
      L[i].joint_w = g(B + 6); L[i].joint_b = g(B + 7);
      L[i].jln_g = g(B + 8); L[i].jln_b = g(B + 9);
      L[i].e1_w = g(B + 10); L[i].e1_b = g(B + 11);
      L[i].e2_w = g(B + 12); L[i].e2_b = g(B + 13);
      L[i].eln_g = g(B + 14); L[i].eln_b = g(B + 15);
      L[i].lat_w = g(B + 16); L[i].lat_b = g(B + 17);
      L[i].nd_w = g(B + 18); L[i].nd_b = g(B + 19);
      L[i].nln_g = g(B + 20); L[i].nln_b = g(B + 21);
    }
  } else {
    return;  // unknown flattening
  }

  // ---- workspace partition ----
  char* base = (char*)d_ws;
  size_t off = 0;
  auto alloc = [&](size_t bytes) -> void* {
    void* p = base + off;
    off = (off + bytes + 255) & ~(size_t)255;
    return p;
  };
  int*   srcIdx = (int*)alloc((size_t)N_EDGE * 4);
  float* nodeB  = (float*)alloc((size_t)N_NODE * 256 * 4);
  float* latB   = (float*)alloc((size_t)N_NODE * 128 * 4);
  float* zB     = (float*)alloc((size_t)N_EDGE * 128 * 4);
  float* ptsB   = (float*)alloc((size_t)N_NODE * 12 * 4);
  float* aggB   = (float*)alloc((size_t)N_NODE * 512 * 4);

  struct WSpec { const float* src; int din, dinPad, dout; _Float16* dst; };
  WSpec specs[34];
  int ns = 0;
  size_t halfs = 0;
  auto countW = [&](const float* src, int din, int dinPad, int dout) {
    specs[ns].src = src; specs[ns].din = din; specs[ns].dinPad = dinPad; specs[ns].dout = dout;
    specs[ns].dst = nullptr; ++ns; halfs += (size_t)dinPad * dout;
  };
  countW(en_w[0], 203, 224, 512);
  countW(en_w[1], 512, 512, 512);
  countW(en_w[2], 512, 512, 256);
  countW(ee_w[0], 416, 416, 256);
  countW(ee_w[1], 256, 256, 256);
  countW(ee_w[2], 256, 256, 128);
  for (int i = 0; i < 4; ++i) {
    countW(L[i].msg1_w, 1168, 1184, 256);
    countW(L[i].msg2_w, 256, 256, 512);
    countW(L[i].joint_w, 512, 512, 512);
    countW(L[i].e1_w, 1152, 1152, 256);
    countW(L[i].e2_w, 256, 256, 128);
    countW(L[i].lat_w, 512, 512, 128);
    countW(L[i].nd_w, 512, 512, 256);
  }
  _Float16* w16 = (_Float16*)alloc(halfs * 2);
  if (off > ws_size) return;  // insufficient workspace
  {
    size_t cur = 0;
    for (int s = 0; s < ns; ++s) {
      specs[s].dst = w16 + cur;
      cur += (size_t)specs[s].dinPad * specs[s].dout;
    }
  }
  _Float16 *fEn0 = specs[0].dst, *fEn1 = specs[1].dst, *fEn2 = specs[2].dst;
  _Float16 *fEe0 = specs[3].dst, *fEe1 = specs[4].dst, *fEe2 = specs[5].dst;
  _Float16 *fM1[4], *fM2[4], *fJ[4], *fE1[4], *fE2[4], *fLa[4], *fNd[4];
  for (int i = 0; i < 4; ++i) {
    fM1[i] = specs[6 + i * 7 + 0].dst;
    fM2[i] = specs[6 + i * 7 + 1].dst;
    fJ[i]  = specs[6 + i * 7 + 2].dst;
    fE1[i] = specs[6 + i * 7 + 3].dst;
    fE2[i] = specs[6 + i * 7 + 4].dst;
    fLa[i] = specs[6 + i * 7 + 5].dst;
    fNd[i] = specs[6 + i * 7 + 6].dst;
  }

  // ---- weight repack (f32 -> f16 fragment order, zero-padded K) ----
  for (int s = 0; s < ns; ++s) {
    const size_t total = (size_t)specs[s].dinPad * specs[s].dout;
    const int blocks = (int)((total + 255) / 256);
    conv_wfrag_kernel<<<blocks, 256, 0, stream>>>(specs[s].src, specs[s].dst,
                                                  specs[s].din, specs[s].dinPad, specs[s].dout);
  }

  // ---- graph build ----
  knn_kernel<<<N_NODE / 128, 128, 0, stream>>>(xca, srcIdx);

  // ---- node embedding ----
  {
    const size_t sm = 32 * 224 * 2 + 32 * 512 * 2 + 32 * 512 * 2 + 32 * 256 * 4;
    node_embed_kernel<<<N_NODE / 32, 256, sm, stream>>>(nsin, mask, fEn0, en_b[0], fEn1, en_b[1],
                                                        fEn2, en_b[2], enln_g, enln_b, nodeB);
  }
  // ---- edge embedding ----
  {
    const size_t sm = 32 * 416 * 2 + 32 * 256 * 2 + 32 * 256 * 2 + 32 * 128 * 4 + 32 * 26 * 4;
    edge_embed_kernel<<<N_EDGE / 32, 256, sm, stream>>>(srcIdx, xca, atoms, mask,
                                                        fEe0, ee_b[0], fEe1, ee_b[1], fEe2, ee_b[2],
                                                        eeln_g, eeln_b, zB);
  }
  hipMemcpyAsync(latB, (void*)lat0, (size_t)N_NODE * 128 * 4, hipMemcpyDeviceToDevice, stream);

  // ---- IPMP layers ----
  const size_t smMsg = 32 * 1184 * 2 + 32 * 256 * 2 + 32 * 16 * 4 + 512 * 4;
  const size_t smEu  = 32 * 1152 * 2 + 32 * 256 * 2 + 32 * 128 * 4;
  const size_t smJ   = 32 * 256 * 4 + 32 * 128 * 4 + 32 * 512 * 2 + 32 * 512 * 4;
  for (int i = 0; i < 4; ++i) {
    points_kernel<<<N_NODE / 8, 128, 0, stream>>>(nodeB, latB, L[i].pt_w, L[i].pt_b, Rm, tv, ptsB);
    msg_kernel<<<N_NODE, 256, smMsg, stream>>>(srcIdx, nodeB, latB, zB, ptsB, Rm,
                                               fM1[i], L[i].msg1_b, fM2[i], L[i].msg2_b, aggB);
    edge_upd_kernel<<<N_EDGE / 32, 256, smEu, stream>>>(srcIdx, nodeB, latB, zB,
                                                        fE1[i], L[i].e1_b, fE2[i], L[i].e2_b,
                                                        L[i].eln_g, L[i].eln_b);
    joint_kernel<<<N_NODE / 32, 256, smJ, stream>>>(aggB, nodeB, latB, mask,
                                                    fJ[i], L[i].joint_b, L[i].jln_g, L[i].jln_b,
                                                    fLa[i], L[i].lat_b,
                                                    fNd[i], L[i].nd_b, L[i].nln_g, L[i].nln_b);
  }
  hipMemcpyAsync(d_out, latB, (size_t)out_size * 4, hipMemcpyDeviceToDevice, stream);
}
